// SelfAttention_4947802325518
// MI455X (gfx1250) — matrix operand order repaired and checked by reading, not compiled, not run on hardware
//
#include <hip/hip_runtime.h>
#include <math.h>

#define UNITS 1024
#define SEQ   4096
#define BATCH 32

#define BM 128
#define BN 64
#define BK 32
#define LDT 40   // padded LDS row stride in bf16 elems (80B, multiple of 16B)

typedef __attribute__((ext_vector_type(16))) __bf16 v16bf;
typedef __attribute__((ext_vector_type(8)))  __bf16 v8bf;
typedef __attribute__((ext_vector_type(8)))  float  v8f;
typedef __attribute__((ext_vector_type(4)))  int    v4i;

#if __has_builtin(__builtin_amdgcn_global_load_async_to_lds_b128) && \
    __has_builtin(__builtin_amdgcn_s_wait_asynccnt)
#define USE_ASYNC_LDS 1
typedef __attribute__((address_space(1))) v4i* gv4i_ptr;
typedef __attribute__((address_space(3))) v4i* lv4i_ptr;
#else
#define USE_ASYNC_LDS 0
#endif

// ---------------- init: zero atomic targets ----------------
__global__ void k_init(float* __restrict__ scores, float* __restrict__ ctx) {
    int i = blockIdx.x * blockDim.x + threadIdx.x;
    if (i < BATCH * SEQ)   scores[i] = 0.0f;
    if (i < BATCH * UNITS) ctx[i]    = 0.0f;
}

// ------- convert U_w [K][N] f32 -> Ut [N][K] bf16 (transposed) -------
__global__ void k_convU(const float* __restrict__ U, __bf16* __restrict__ Ut) {
    int n = blockIdx.x;  // 0..1023
    for (int k = threadIdx.x; k < UNITS; k += blockDim.x)
        Ut[n * UNITS + k] = (__bf16)U[k * UNITS + n];
}

// ---------------- w_s = s_prev @ W_w + W_b ----------------
__global__ void k_ws(const float* __restrict__ s_prev, const float* __restrict__ W,
                     const float* __restrict__ Wb, float* __restrict__ wsout) {
    int b = blockIdx.y;
    int j = blockIdx.x * blockDim.x + threadIdx.x;
    float acc = Wb[j];
    const float* sp = s_prev + b * UNITS;
    for (int k = 0; k < UNITS; ++k)
        acc += sp[k] * W[k * UNITS + j];
    wsout[b * UNITS + j] = acc;
}

// epilogue helper: score partial = sum_j tanh(acc + w_s[j]+U_b[j]) * V_w[j]
__device__ __forceinline__ void epilogue(v8f acc, int j, int half, int jc,
                                         const float* __restrict__ wsb,
                                         const float* __restrict__ Ub,
                                         const float* __restrict__ Vw,
                                         float* __restrict__ scores, long ibase) {
    float c  = wsb[j] + Ub[j];
    float vw = Vw[j];
#pragma unroll
    for (int r = 0; r < 8; ++r) {
        float v = tanhf(acc[r] + c) * vw;
        v += __shfl_xor(v, 1);
        v += __shfl_xor(v, 2);
        v += __shfl_xor(v, 4);
        v += __shfl_xor(v, 8);   // lanes of same half now hold 16-col sum
        if (jc == 0)
            atomicAdd(&scores[ibase + r + half * 8], v);
    }
}

// ------- fused: scores[b,s] += sum_j tanh(w_s + H@U_w + U_b)*V_w -------
__global__ __launch_bounds__(256)
void k_scores(const float* __restrict__ H,     // [B*S, U] f32
              const __bf16* __restrict__ Ut,   // [N=U][K=U] bf16
              const float* __restrict__ wsb_g, // [B, U]
              const float* __restrict__ Ub, const float* __restrict__ Vw,
              float* __restrict__ scores)      // [B*S]
{
    __shared__ __attribute__((aligned(16))) __bf16 As[2][BM][LDT];
    __shared__ __attribute__((aligned(16))) __bf16 Bs[2][BN][LDT];

    const int tid   = threadIdx.x;
    const int lane  = tid & 31;
    const int wave  = tid >> 5;      // 8 waves
    const int waveM = wave >> 1;     // 0..3 : 32-row strip
    const int waveN = wave & 1;      // 0..1 : 32-col strip

    const int  n0 = blockIdx.x * BN;
    const long i0 = (long)blockIdx.y * BM;   // row block (stays within one batch)
    const int  b  = (int)(i0 >> 12);         // 4096 rows per batch

    // A staging: 128 rows x 32 K f32 -> bf16 ; thread = (row, 16-elem half)
    const int arow  = tid >> 1;
    const int ahalf = (tid & 1) * 16;
    // B staging: 64 rows x 32 K bf16 ; thread = (row, 8-elem seg)
    const int brow  = tid >> 2;
    const int bseg  = (tid & 3) * 8;

    auto stageA = [&](int bufi, int k0) {
        const float* src = H + (i0 + arow) * UNITS + k0 + ahalf;
        float4 f0 = *(const float4*)(src);
        float4 f1 = *(const float4*)(src + 4);
        float4 f2 = *(const float4*)(src + 8);
        float4 f3 = *(const float4*)(src + 12);
        v8bf lo, hi;
        lo[0] = (__bf16)f0.x; lo[1] = (__bf16)f0.y;
        lo[2] = (__bf16)f0.z; lo[3] = (__bf16)f0.w;
        lo[4] = (__bf16)f1.x; lo[5] = (__bf16)f1.y;
        lo[6] = (__bf16)f1.z; lo[7] = (__bf16)f1.w;
        hi[0] = (__bf16)f2.x; hi[1] = (__bf16)f2.y;
        hi[2] = (__bf16)f2.z; hi[3] = (__bf16)f2.w;
        hi[4] = (__bf16)f3.x; hi[5] = (__bf16)f3.y;
        hi[6] = (__bf16)f3.z; hi[7] = (__bf16)f3.w;
        *(v8bf*)&As[bufi][arow][ahalf]     = lo;
        *(v8bf*)&As[bufi][arow][ahalf + 8] = hi;
        __builtin_prefetch(src + BK, 0, 3);      // global_prefetch_b8, near scope
    };

    auto stageB = [&](int bufi, int k0) {
        const __bf16* src = Ut + (long)(n0 + brow) * UNITS + k0 + bseg;
        __bf16* dst = &Bs[bufi][brow][bseg];
#if USE_ASYNC_LDS
        __builtin_amdgcn_global_load_async_to_lds_b128(
            (gv4i_ptr)(v4i*)(uintptr_t)src,   // global addrspace(1) v4i*
            (lv4i_ptr)dst,                    // LDS addrspace(3) v4i*
            0, 0);
#else
        *(v8bf*)dst = *(const v8bf*)src;
#endif
    };

    v8f acc00 = {}, acc01 = {}, acc10 = {}, acc11 = {};

    const int mlane = lane & 15;
    const int kbA   = (lane < 16) ? 0 : 8;    // A: K {0..7,16..23} / {8..15,24..31}
    const int kbB   = (lane < 16) ? 0 : 8;    // B: K {0..7,16..23} / {8..15,24..31}
    const int arow0 = waveM * 32 + mlane;
    const int brow0 = waveN * 32 + mlane;

    stageA(0, 0);
    stageB(0, 0);

    int buf = 0;
    for (int k0 = 0; k0 < UNITS; k0 += BK) {
#if USE_ASYNC_LDS
        __builtin_amdgcn_s_wait_asynccnt(0);
#endif
        __syncthreads();                       // staged tile `buf` visible

        if (k0 + BK < UNITS) {                 // stage next tile into other buffer
            stageA(buf ^ 1, k0 + BK);
            stageB(buf ^ 1, k0 + BK);
        }

        union { v16bf v; v8bf h[2]; } A0, A1, B0, B1;
        A0.h[0] = *(const v8bf*)&As[buf][arow0][kbA];
        A0.h[1] = *(const v8bf*)&As[buf][arow0][kbA + 16];
        A1.h[0] = *(const v8bf*)&As[buf][arow0 + 16][kbA];
        A1.h[1] = *(const v8bf*)&As[buf][arow0 + 16][kbA + 16];
        B0.h[0] = *(const v8bf*)&Bs[buf][brow0][kbB];
        B0.h[1] = *(const v8bf*)&Bs[buf][brow0][kbB + 16];
        B1.h[0] = *(const v8bf*)&Bs[buf][brow0 + 16][kbB];
        B1.h[1] = *(const v8bf*)&Bs[buf][brow0 + 16][kbB + 16];

        acc00 = __builtin_amdgcn_wmma_f32_16x16x32_bf16(
                    false, A0.v, false, B0.v, (short)0, acc00, false, false);
        acc01 = __builtin_amdgcn_wmma_f32_16x16x32_bf16(
                    false, A0.v, false, B1.v, (short)0, acc01, false, false);
        acc10 = __builtin_amdgcn_wmma_f32_16x16x32_bf16(
                    false, A1.v, false, B0.v, (short)0, acc10, false, false);
        acc11 = __builtin_amdgcn_wmma_f32_16x16x32_bf16(
                    false, A1.v, false, B1.v, (short)0, acc11, false, false);

        buf ^= 1;
    }

    // epilogue: C layout -> lane half selects M=r or M=r+8, N = lane%16
    const int  jc  = lane & 15;
    const int  half = lane >> 4;
    const long ib0 = i0 + waveM * 32;
    const int  j0  = n0 + waveN * 32 + jc;
    const float* wsb = wsb_g + (long)b * UNITS;
    epilogue(acc00, j0,      half, jc, wsb, Ub, Vw, scores, ib0);
    epilogue(acc01, j0 + 16, half, jc, wsb, Ub, Vw, scores, ib0);
    epilogue(acc10, j0,      half, jc, wsb, Ub, Vw, scores, ib0 + 16);
    epilogue(acc11, j0 + 16, half, jc, wsb, Ub, Vw, scores, ib0 + 16);
}

// ---------------- softmax over S per batch ----------------
__global__ void k_softmax(const float* __restrict__ scores, float* __restrict__ wout) {
    __shared__ float red[256];
    const int b = blockIdx.x;
    const float* s = scores + (long)b * SEQ;

    float m = -1e30f;
    for (int i = threadIdx.x; i < SEQ; i += 256) m = fmaxf(m, s[i]);
    red[threadIdx.x] = m; __syncthreads();
    for (int o = 128; o > 0; o >>= 1) {
        if (threadIdx.x < o) red[threadIdx.x] = fmaxf(red[threadIdx.x], red[threadIdx.x + o]);
        __syncthreads();
    }
    m = red[0]; __syncthreads();

    float sum = 0.0f;
    for (int i = threadIdx.x; i < SEQ; i += 256) sum += __expf(s[i] - m);
    red[threadIdx.x] = sum; __syncthreads();
    for (int o = 128; o > 0; o >>= 1) {
        if (threadIdx.x < o) red[threadIdx.x] += red[threadIdx.x + o];
        __syncthreads();
    }
    const float inv = 1.0f / red[0];

    float* w = wout + (long)b * SEQ;
    for (int i = threadIdx.x; i < SEQ; i += 256) w[i] = __expf(s[i] - m) * inv;
}

// ---------------- context = sum_s weights * hidden (float4) ----------------
__global__ void k_context(const float* __restrict__ H, const float* __restrict__ w,
                          float* __restrict__ ctx) {
    const int b  = blockIdx.z;
    const int u4 = threadIdx.x;            // 256 float4 per row
    const int s0 = blockIdx.y * 256;       // grid.y = 16 s-chunks
    const float* wb = w + (long)b * SEQ;
    float4 acc = {0.f, 0.f, 0.f, 0.f};
    for (int s = s0; s < s0 + 256; ++s) {
        const float ws = wb[s];
        float4 h = *(const float4*)(H + ((long)b * SEQ + s) * UNITS + u4 * 4);
        acc.x += ws * h.x; acc.y += ws * h.y;
        acc.z += ws * h.z; acc.w += ws * h.w;
    }
    float* c = ctx + b * UNITS + u4 * 4;
    atomicAdd(c + 0, acc.x);
    atomicAdd(c + 1, acc.y);
    atomicAdd(c + 2, acc.z);
    atomicAdd(c + 3, acc.w);
}

extern "C" void kernel_launch(void* const* d_in, const int* in_sizes, int n_in,
                              void* d_out, int out_size, void* d_ws, size_t ws_size,
                              hipStream_t stream) {
    const float* s_prev = (const float*)d_in[0];
    const float* H      = (const float*)d_in[1];
    const float* W_w    = (const float*)d_in[2];
    const float* W_b    = (const float*)d_in[3];
    const float* U_w    = (const float*)d_in[4];
    const float* U_b    = (const float*)d_in[5];
    const float* V_w    = (const float*)d_in[6];
    // d_in[7] = V_b: constant shift, cancels in softmax -> unused

    float* ctx  = (float*)d_out;                    // [32,1024]
    float* wout = (float*)d_out + BATCH * UNITS;    // [32,4096]

    float*  scores = (float*)d_ws;                  // 131072 f32
    float*  wsbuf  = scores + BATCH * SEQ;          // 32768  f32
    __bf16* Ut     = (__bf16*)(wsbuf + BATCH * UNITS); // 1M bf16 (~2MB)

    k_init   <<<dim3(512),       256, 0, stream>>>(scores, ctx);
    k_convU  <<<dim3(UNITS),     256, 0, stream>>>(U_w, Ut);
    k_ws     <<<dim3(4, BATCH),  256, 0, stream>>>(s_prev, W_w, W_b, wsbuf);
    k_scores <<<dim3(UNITS / BN, (BATCH * SEQ) / BM), 256, 0, stream>>>(
                 H, Ut, wsbuf, U_b, V_w, scores);
    k_softmax<<<dim3(BATCH),     256, 0, stream>>>(scores, wout);
    k_context<<<dim3(1, 16, BATCH), 256, 0, stream>>>(H, wout, ctx);
}
